// MultiHeadAttention_61469571940468
// MI455X (gfx1250) — matrix operand order repaired and checked by reading, not compiled, not run on hardware
//
#include <hip/hip_runtime.h>
#include <hip/hip_bf16.h>

#define SEQ   4096
#define EMB   1024
#define HEADS 16
#define DH    128
#define HV    (HEADS * DH)   // 2048

typedef __bf16 bf16_t;
typedef __attribute__((ext_vector_type(16))) __bf16 v16bf;
typedef __attribute__((ext_vector_type(8)))  __bf16 v8bf;
typedef __attribute__((ext_vector_type(8)))  float  v8f;

typedef unsigned int u32x4 __attribute__((ext_vector_type(4)));
typedef int          i32x8 __attribute__((ext_vector_type(8)));
typedef int          i32x4 __attribute__((ext_vector_type(4)));

union ABu { v16bf v; v8bf h[2]; };

__device__ __forceinline__ v8f wmma_bf16(v16bf a, v16bf b, v8f c) {
  return __builtin_amdgcn_wmma_f32_16x16x32_bf16(
      false, a, false, b, (short)0, c, false, false);
}

// ---- Tensor Data Mover: 2D tile (tile1 rows of tile0 8-byte units) -> LDS ----
// stride0 = row stride in 8-byte units. tensor dims == tile dims (no OOB clip).
__device__ __forceinline__ void tdm_load_2d(const void* gptr, void* lptr,
                                            unsigned tile0, unsigned tile1,
                                            unsigned long long stride0) {
  unsigned long long ga = (unsigned long long)(uintptr_t)gptr;
  unsigned lds = (unsigned)(uintptr_t)lptr;   // low 32 bits = LDS byte address
  u32x4 g0;
  g0[0] = 1u;                                           // count=1, user mode
  g0[1] = lds;                                          // lds_addr
  g0[2] = (unsigned)ga;                                 // global_addr[31:0]
  g0[3] = ((unsigned)(ga >> 32) & 0x01FFFFFFu) | (2u << 30); // addr[56:32]|type=2
  i32x8 g1;
  g1[0] = (int)(3u << 16);                              // data_size = 8 bytes
  g1[1] = (int)((tile0 & 0xFFFFu) << 16);               // tensor_dim0[15:0]
  g1[2] = (int)((tile0 >> 16) | ((tile1 & 0xFFFFu) << 16)); // td0 hi | td1 lo
  g1[3] = (int)((tile1 >> 16) | (tile0 << 16));         // td1 hi | tile_dim0
  g1[4] = (int)(tile1 & 0xFFFFu);                       // tile_dim1 (tile_dim2=0)
  g1[5] = (int)(unsigned)stride0;                       // tensor_dim0_stride lo
  g1[6] = (int)((unsigned)(stride0 >> 32) & 0xFFFFu);   // stride hi (dim1_stride=0)
  g1[7] = 0;
  i32x4 gz4 = {0, 0, 0, 0};
  i32x8 gz8 = {0, 0, 0, 0, 0, 0, 0, 0};
  // clang-23 / therock-10.0 form: 6 args (g0, g1, g2, g3, g4, cpol)
  __builtin_amdgcn_tensor_load_to_lds(g0, g1, gz4, gz4, gz8, 0);
}

// max/sum over the 16 lanes of a half-wave (xor masks stay inside each half)
__device__ __forceinline__ float halfmax16(float x) {
  x = fmaxf(x, __shfl_xor(x, 1, 32));
  x = fmaxf(x, __shfl_xor(x, 2, 32));
  x = fmaxf(x, __shfl_xor(x, 4, 32));
  x = fmaxf(x, __shfl_xor(x, 8, 32));
  return x;
}
__device__ __forceinline__ float halfsum16(float x) {
  x += __shfl_xor(x, 1, 32);
  x += __shfl_xor(x, 2, 32);
  x += __shfl_xor(x, 4, 32);
  x += __shfl_xor(x, 8, 32);
  return x;
}

// ---------------- K0: xp = x + positional encoding, store bf16 ----------------
__global__ void xp_kernel(const float* __restrict__ x, bf16_t* __restrict__ xp) {
  int s = blockIdx.x;
  int t = threadIdx.x;
  float fs = (float)s;
#pragma unroll
  for (int q = 0; q < 4; ++q) {
    int e = t * 4 + q;
    int i = e >> 1;
    float ang = fs * 1024.0f * __powf(10000.0f, -(float)(2 * i));
    float pe = (e & 1) ? cosf(ang) : sinf(ang);
    xp[(size_t)s * EMB + e] = (bf16_t)(x[(size_t)s * EMB + e] + pe);
  }
}

// ------------- K1: convert weights to bf16 and transpose -------------
__global__ void conv_kernel(const float* __restrict__ wq, const float* __restrict__ wk,
                            const float* __restrict__ wv, const float* __restrict__ wo,
                            bf16_t* __restrict__ wqt, bf16_t* __restrict__ wkt,
                            bf16_t* __restrict__ wvt, bf16_t* __restrict__ wot) {
  size_t tid = (size_t)blockIdx.x * blockDim.x + threadIdx.x;
  const size_t NW = (size_t)HEADS * EMB * DH;
  if (tid < 3 * NW) {
    int z = (int)(tid / NW);
    size_t r = tid % NW;
    int h  = (int)(r / ((size_t)EMB * DH));
    int rr = (int)(r % ((size_t)EMB * DH));
    int e = rr / DH, d = rr % DH;
    const float* src = (z == 0) ? wq : (z == 1) ? wk : wv;
    bf16_t*      dst = (z == 0) ? wqt : (z == 1) ? wkt : wvt;
    dst[((size_t)h * DH + d) * EMB + e] = (bf16_t)src[((size_t)h * EMB + e) * DH + d];
  } else if (tid < 3 * NW + (size_t)HV * EMB) {
    size_t r = tid - 3 * NW;
    int row = (int)(r / EMB), col = (int)(r % EMB);
    wot[(size_t)col * HV + row] = (bf16_t)wo[(size_t)row * EMB + col];
  }
}

// ------------- K2: Q/K/V projections via WMMA -------------
__global__ void __launch_bounds__(128)
qkv_kernel(const bf16_t* __restrict__ xp, const bf16_t* __restrict__ wt_base,
           const float* __restrict__ bq, const float* __restrict__ bk,
           const float* __restrict__ bv,
           bf16_t* __restrict__ q, bf16_t* __restrict__ k, bf16_t* __restrict__ vt) {
  int rb = blockIdx.x, h = blockIdx.y, z = blockIdx.z;
  int tid = threadIdx.x, w = tid >> 5, lane = tid & 31;
  int m = lane & 15, g = lane >> 4;
  int rowbase = rb * 64 + w * 16;

  const bf16_t* Wt = wt_base + (size_t)z * HEADS * DH * EMB + (size_t)h * DH * EMB;
  const float* bias = (z == 0) ? bq : (z == 1) ? bk : bv;

  __shared__ __align__(16) bf16_t Bsh[DH * 32];   // 128 rows (n) x 32 (k)

  v8f acc[8] = {};

  for (int kc = 0; kc < EMB; kc += 32) {
    __syncthreads();
    if (w == 0) {
      // B tile: 128 rows x 64B, row stride EMB*2 bytes (=256 x 8B units)
      tdm_load_2d(&Wt[kc], Bsh, 8u, 128u, (unsigned long long)(EMB * 2 / 8));
      __builtin_amdgcn_s_wait_tensorcnt(0);
    }
    __syncthreads();

    ABu a;
    const bf16_t* arow = xp + (size_t)(rowbase + m) * EMB + kc;
    a.h[0] = *(const v8bf*)(arow + g * 16);
    a.h[1] = *(const v8bf*)(arow + g * 16 + 8);

    ABu b[8];
#pragma unroll
    for (int n = 0; n < 8; ++n) {
      const bf16_t* brow = &Bsh[(n * 16 + m) * 32 + g * 16];
      b[n].h[0] = *(const v8bf*)(brow);
      b[n].h[1] = *(const v8bf*)(brow + 8);
    }
#pragma unroll
    for (int n = 0; n < 8; ++n) acc[n] = wmma_bf16(a.v, b[n].v, acc[n]);
  }

#pragma unroll
  for (int n = 0; n < 8; ++n) {
    int col = n * 16 + m;
    float bc = bias[h * DH + col];
#pragma unroll
    for (int r = 0; r < 8; ++r) {
      int row = rowbase + r + 8 * g;
      bf16_t o = (bf16_t)(acc[n][r] + bc);
      if (z == 0)      q[((size_t)h * SEQ + row) * DH + col] = o;
      else if (z == 1) k[((size_t)h * SEQ + row) * DH + col] = o;
      else             vt[((size_t)h * DH + col) * SEQ + row] = o;
    }
  }
}

// ------------- K3: flash attention (online softmax) -------------
__global__ void __launch_bounds__(128)
attn_kernel(const bf16_t* __restrict__ Q, const bf16_t* __restrict__ K,
            const bf16_t* __restrict__ VT, bf16_t* __restrict__ CAT) {
  int rb = blockIdx.x, h = blockIdx.y;
  int tid = threadIdx.x, w = tid >> 5, lane = tid & 31;
  int m = lane & 15, g = lane >> 4;
  int rowbase = rb * 64 + w * 16;

  __shared__ __align__(16) bf16_t Ksh[32 * DH];      // [key][feat]
  __shared__ __align__(16) bf16_t Vsh[DH * 32];      // [feat][key]
  __shared__ __align__(16) bf16_t Psh[4][16 * 32];   // per-wave P

  ABu aq[4];
  const bf16_t* qrow = Q + ((size_t)h * SEQ + rowbase + m) * DH;
#pragma unroll
  for (int c = 0; c < 4; ++c) {
    aq[c].h[0] = *(const v8bf*)(qrow + c * 32 + g * 16);
    aq[c].h[1] = *(const v8bf*)(qrow + c * 32 + g * 16 + 8);
  }

  v8f O[8] = {};
  float mrow[8], lrow[8];
#pragma unroll
  for (int r = 0; r < 8; ++r) { mrow[r] = -1e30f; lrow[r] = 0.0f; }
  const float c0 = 0.08838834764831845f * 1.4426950408889634f;  // scale*log2(e)

  for (int j = 0; j < SEQ; j += 32) {
    __syncthreads();
    if (w == 0) {
      // K tile: 32 rows x 256B contiguous = one 8KB line (1024 x 8B units)
      tdm_load_2d(&K[((size_t)h * SEQ + j) * DH], Ksh, 1024u, 1u, 1024ull);
      // V tile: 128 rows x 64B, row stride SEQ*2 bytes (=1024 x 8B units)
      tdm_load_2d(&VT[(size_t)h * DH * SEQ + j], Vsh, 8u, 128u, 1024ull);
      __builtin_amdgcn_s_wait_tensorcnt(0);
    }
    __syncthreads();

    // scores: two 16x16 tiles over K=128 (4 chunks) -- preload all B operands
    ABu bk_[8];
#pragma unroll
    for (int c = 0; c < 4; ++c) {
      const bf16_t* kr0 = &Ksh[m * DH + c * 32 + g * 16];
      const bf16_t* kr1 = &Ksh[(16 + m) * DH + c * 32 + g * 16];
      bk_[c].h[0]     = *(const v8bf*)kr0;  bk_[c].h[1]     = *(const v8bf*)(kr0 + 8);
      bk_[4 + c].h[0] = *(const v8bf*)kr1;  bk_[4 + c].h[1] = *(const v8bf*)(kr1 + 8);
    }
    v8f s0 = {}, s1 = {};
#pragma unroll
    for (int c = 0; c < 4; ++c) {
      s0 = wmma_bf16(aq[c].v, bk_[c].v, s0);
      s1 = wmma_bf16(aq[c].v, bk_[4 + c].v, s1);
    }

    // online softmax update (row r+8g lives in this half-wave)
    float alpha[8];
#pragma unroll
    for (int r = 0; r < 8; ++r) {
      float tmax = halfmax16(fmaxf(s0[r], s1[r]));
      float mn = fmaxf(mrow[r], tmax);
      alpha[r] = exp2f((mrow[r] - mn) * c0);
      mrow[r] = mn;
      float p0 = exp2f((s0[r] - mn) * c0);
      float p1 = exp2f((s1[r] - mn) * c0);
      int row = r + 8 * g;
      Psh[w][row * 32 + m]      = (bf16_t)p0;
      Psh[w][row * 32 + 16 + m] = (bf16_t)p1;
      lrow[r] = lrow[r] * alpha[r] + halfsum16(p0 + p1);
    }
#pragma unroll
    for (int n = 0; n < 8; ++n)
#pragma unroll
      for (int r = 0; r < 8; ++r) O[n][r] *= alpha[r];

    // P (16x32, A layout from LDS) x V (32x128) -- preload all B operands
    ABu ap;
    ap.h[0] = *(const v8bf*)&Psh[w][m * 32 + g * 16];
    ap.h[1] = *(const v8bf*)&Psh[w][m * 32 + g * 16 + 8];
    ABu bv_[8];
#pragma unroll
    for (int n = 0; n < 8; ++n) {
      const bf16_t* vr = &Vsh[(n * 16 + m) * 32 + g * 16];
      bv_[n].h[0] = *(const v8bf*)vr; bv_[n].h[1] = *(const v8bf*)(vr + 8);
    }
#pragma unroll
    for (int n = 0; n < 8; ++n) O[n] = wmma_bf16(ap.v, bv_[n].v, O[n]);
  }

#pragma unroll
  for (int n = 0; n < 8; ++n) {
    int col = h * DH + n * 16 + m;
#pragma unroll
    for (int r = 0; r < 8; ++r) {
      int row = rowbase + r + 8 * g;
      CAT[(size_t)row * HV + col] = (bf16_t)(O[n][r] / lrow[r]);
    }
  }
}

// ------------- K4: output projection  out = cat @ Wo + bo -------------
__global__ void __launch_bounds__(128)
outproj_kernel(const bf16_t* __restrict__ cat, const bf16_t* __restrict__ wot,
               const float* __restrict__ bo, float* __restrict__ out) {
  int rb = blockIdx.x, nb = blockIdx.y;
  int tid = threadIdx.x, w = tid >> 5, lane = tid & 31;
  int m = lane & 15, g = lane >> 4;
  int rowbase = rb * 64 + w * 16;

  __shared__ __align__(16) bf16_t Bsh[128 * 32];

  v8f acc[8] = {};
  for (int kc = 0; kc < HV; kc += 32) {
    __syncthreads();
    if (w == 0) {
      // B tile: 128 rows x 64B, row stride HV*2 bytes (=512 x 8B units)
      tdm_load_2d(&wot[(size_t)(nb * 128) * HV + kc], Bsh, 8u, 128u,
                  (unsigned long long)(HV * 2 / 8));
      __builtin_amdgcn_s_wait_tensorcnt(0);
    }
    __syncthreads();

    ABu a;
    const bf16_t* ar = cat + (size_t)(rowbase + m) * HV + kc;
    a.h[0] = *(const v8bf*)(ar + g * 16);
    a.h[1] = *(const v8bf*)(ar + g * 16 + 8);

    ABu b[8];
#pragma unroll
    for (int n = 0; n < 8; ++n) {
      const bf16_t* brow = &Bsh[(n * 16 + m) * 32 + g * 16];
      b[n].h[0] = *(const v8bf*)brow; b[n].h[1] = *(const v8bf*)(brow + 8);
    }
#pragma unroll
    for (int n = 0; n < 8; ++n) acc[n] = wmma_bf16(a.v, b[n].v, acc[n]);
  }

#pragma unroll
  for (int n = 0; n < 8; ++n) {
    int col = nb * 128 + n * 16 + m;
    float bc = bo[col];
#pragma unroll
    for (int r = 0; r < 8; ++r) {
      int row = rowbase + r + 8 * g;
      out[(size_t)row * EMB + col] = acc[n][r] + bc;
    }
  }
}

extern "C" void kernel_launch(void* const* d_in, const int* in_sizes, int n_in,
                              void* d_out, int out_size, void* d_ws, size_t ws_size,
                              hipStream_t stream) {
  const float* x  = (const float*)d_in[0];
  const float* Wq = (const float*)d_in[1];
  const float* bq = (const float*)d_in[2];
  const float* Wk = (const float*)d_in[3];
  const float* bk = (const float*)d_in[4];
  const float* Wv = (const float*)d_in[5];
  const float* bv = (const float*)d_in[6];
  const float* Wo = (const float*)d_in[7];
  const float* bo = (const float*)d_in[8];
  float* out = (float*)d_out;

  char* ws = (char*)d_ws;
  const size_t XP_OFF  = 0;
  const size_t WQT_OFF = XP_OFF  + (size_t)SEQ * EMB * 2;
  const size_t WKT_OFF = WQT_OFF + (size_t)HEADS * DH * EMB * 2;
  const size_t WVT_OFF = WKT_OFF + (size_t)HEADS * DH * EMB * 2;
  const size_t WOT_OFF = WVT_OFF + (size_t)HEADS * DH * EMB * 2;
  const size_t Q_OFF   = WOT_OFF + (size_t)HV * EMB * 2;
  const size_t K_OFF   = Q_OFF   + (size_t)HEADS * SEQ * DH * 2;
  const size_t VT_OFF  = K_OFF   + (size_t)HEADS * SEQ * DH * 2;
  const size_t CAT_OFF = VT_OFF  + (size_t)HEADS * SEQ * DH * 2;

  bf16_t* xp  = (bf16_t*)(ws + XP_OFF);
  bf16_t* wqt = (bf16_t*)(ws + WQT_OFF);
  bf16_t* wkt = (bf16_t*)(ws + WKT_OFF);
  bf16_t* wvt = (bf16_t*)(ws + WVT_OFF);
  bf16_t* wot = (bf16_t*)(ws + WOT_OFF);
  bf16_t* Qb  = (bf16_t*)(ws + Q_OFF);
  bf16_t* Kb  = (bf16_t*)(ws + K_OFF);
  bf16_t* VTb = (bf16_t*)(ws + VT_OFF);
  bf16_t* CATb= (bf16_t*)(ws + CAT_OFF);

  xp_kernel<<<SEQ, 256, 0, stream>>>(x, xp);
  conv_kernel<<<32768, 256, 0, stream>>>(Wq, Wk, Wv, Wo, wqt, wkt, wvt, wot);
  qkv_kernel<<<dim3(SEQ / 64, HEADS, 3), 128, 0, stream>>>(xp, wqt, bq, bk, bv,
                                                           Qb, Kb, VTb);
  attn_kernel<<<dim3(SEQ / 64, HEADS), 128, 0, stream>>>(Qb, Kb, VTb, CATb);
  outproj_kernel<<<dim3(SEQ / 64, EMB / 128), 128, 0, stream>>>(CATb, wot, bo, out);
}
